// HandGNN_40776419508499
// MI455X (gfx1250) — matrix operand order repaired and checked by reading, not compiled, not run on hardware
//
#include <hip/hip_runtime.h>
#include <hip/hip_bf16.h>

#define NN 262144      // nodes
#define NE 4194304     // edges
#define NG 16384       // graphs
#define NC 26          // classes

typedef __attribute__((ext_vector_type(16))) __bf16 v16bf;
typedef __attribute__((ext_vector_type(8)))  float  v8f;

union BF16x16 { v16bf v; unsigned short u[16]; };

static __device__ __forceinline__ unsigned short f2bf(float f) {
    unsigned u = __float_as_uint(f);
    u = u + 0x7FFFu + ((u >> 16) & 1u);   // round-to-nearest-even
    return (unsigned short)(u >> 16);
}

// ---------------- degree + norm ----------------
__global__ void k_deg(const int* __restrict__ dst, float* __restrict__ deg) {
    int e = blockIdx.x * blockDim.x + threadIdx.x;
    if (e < NE) atomicAdd(&deg[dst[e]], 1.0f);
}

__global__ void k_dis(float* __restrict__ d) {
    int i = blockIdx.x * blockDim.x + threadIdx.x;
    if (i < NN) d[i] = rsqrtf(d[i] + 2.0f);   // improved=True: +2 self loop
}

// ---------------- layer 1: x[N,3] @ W1[3,16] ----------------
__global__ void k_xw1(const float* __restrict__ x, const float* __restrict__ W1,
                      float* __restrict__ h1lin) {
    int i = blockIdx.x * blockDim.x + threadIdx.x;
    if (i >= NN) return;
    float x0 = x[i * 3 + 0], x1 = x[i * 3 + 1], x2 = x[i * 3 + 2];
#pragma unroll
    for (int j = 0; j < 16; ++j)
        h1lin[(size_t)i * 16 + j] =
            fmaf(x0, W1[j], fmaf(x1, W1[16 + j], x2 * W1[32 + j]));
}

// ---------------- edge scatter: thread per (edge, feature) ----------------
template <int LOGF>
__global__ void k_scatter(const int* __restrict__ src, const int* __restrict__ dst,
                          const float* __restrict__ dis, const float* __restrict__ h,
                          float* __restrict__ agg) {
    const int F = 1 << LOGF;
    unsigned tid = blockIdx.x * blockDim.x + threadIdx.x;
    int e = tid >> LOGF, j = tid & (F - 1);
    int s = src[e], d = dst[e];
    float nrm = dis[s] * dis[d];
    atomicAdd(&agg[(size_t)d * F + j], nrm * h[(size_t)s * F + j]);
}

// ---------------- post layer 1: self-loop + bias + relu -> bf16 ----------------
__global__ void k_post1(const float* __restrict__ agg1, const float* __restrict__ h1lin,
                        const float* __restrict__ dis, const float* __restrict__ b1,
                        unsigned short* __restrict__ h1bf) {
    unsigned tid = blockIdx.x * blockDim.x + threadIdx.x;  // N*16
    int i = tid >> 4, j = tid & 15;
    float s = dis[i];
    float v = agg1[tid] + 2.0f * s * s * h1lin[tid] + b1[j];
    h1bf[tid] = f2bf(fmaxf(v, 0.0f));
}

// ---------------- layer 2 GEMM: h1[N,16] @ W2[16,32] via WMMA bf16 ----------------
// K=16 padded to 32 with zeros. Each wave: 4 M-tiles of 16 rows, 2 N-tiles of 16 cols.
// Also zero-fills agg2 (same addresses as h2lin stores).
#define TPW 4
__global__ void k_gemm2(const unsigned short* __restrict__ h1bf,
                        const float* __restrict__ W2,
                        float* __restrict__ h2lin, float* __restrict__ agg2) {
    const int lane = threadIdx.x & 31;
    const int wave = blockIdx.x * (blockDim.x >> 5) + (threadIdx.x >> 5);
    const int n = lane & 15;
    const int hiHalf = lane >> 4;

    // B tiles: lanes 0-15 hold K=0..15 (real rows of W2), lanes 16-31 hold K=16..31 (zero pad)
    BF16x16 b0, b1;
#pragma unroll
    for (int k = 0; k < 16; ++k) {
        unsigned short w0 = 0, w1 = 0;
        if (lane < 16) {
            w0 = f2bf(W2[k * 32 + n]);
            w1 = f2bf(W2[k * 32 + 16 + n]);
        }
        b0.u[k] = w0; b1.u[k] = w1;
    }

#pragma unroll
    for (int t = 0; t < TPW; ++t) {
        const int tile = wave * TPW + t;
        const int row = tile * 16 + n;
        // A tile: lanes 0-15 -> K 0..15 in u[0..15]; lanes 16-31 -> K 16..31 (zero pad)
        BF16x16 a;
        const unsigned short* ap = h1bf + (size_t)row * 16;
#pragma unroll
        for (int q = 0; q < 8; ++q) { a.u[q] = hiHalf ? 0 : ap[q]; a.u[8 + q] = hiHalf ? 0 : ap[8 + q]; }

        v8f c0 = {}, c1 = {};
        c0 = __builtin_amdgcn_wmma_f32_16x16x32_bf16(false, a.v, false, b0.v,
                                                     (short)0, c0, false, false);
        c1 = __builtin_amdgcn_wmma_f32_16x16x32_bf16(false, a.v, false, b1.v,
                                                     (short)0, c1, false, false);
        // D layout: lane holds column n; VGPR r = row r (lanes 0-15) / row r+8 (lanes 16-31)
#pragma unroll
        for (int r = 0; r < 8; ++r) {
            const size_t orow = (size_t)tile * 16 + r + hiHalf * 8;
            h2lin[orow * 32 + n]      = c0[r];
            h2lin[orow * 32 + 16 + n] = c1[r];
            agg2[orow * 32 + n]       = 0.0f;
            agg2[orow * 32 + 16 + n]  = 0.0f;
        }
    }
}

// ---------------- post layer 2: self-loop + bias + relu + pooled atomics ----------------
__global__ void k_post2(const float* __restrict__ agg2, const float* __restrict__ h2lin,
                        const float* __restrict__ dis, const float* __restrict__ b2,
                        const int* __restrict__ batch,
                        float* __restrict__ pooled, float* __restrict__ cnt) {
    unsigned tid = blockIdx.x * blockDim.x + threadIdx.x;  // N*32
    int i = tid >> 5, j = tid & 31;
    float s = dis[i];
    float v = agg2[tid] + 2.0f * s * s * h2lin[tid] + b2[j];
    v = fmaxf(v, 0.0f);
    int g = batch[i];
    atomicAdd(&pooled[(size_t)g * 32 + j], v);
    if (j == 0) atomicAdd(&cnt[g], 1.0f);
}

__global__ void k_pooldiv(const float* __restrict__ pooled, const float* __restrict__ cnt,
                          unsigned short* __restrict__ pooledbf) {
    unsigned tid = blockIdx.x * blockDim.x + threadIdx.x;  // G*32
    int g = tid >> 5;
    pooledbf[tid] = f2bf(pooled[tid] / fmaxf(cnt[g], 1.0f));
}

// ---------------- classifier: pooled[G,32] @ Wl[32,26] + bl via WMMA bf16 ----------------
__global__ void k_cls(const unsigned short* __restrict__ pooledbf,
                      const float* __restrict__ Wl, const float* __restrict__ bl,
                      float* __restrict__ out) {
    const int lane = threadIdx.x & 31;
    const int tile = blockIdx.x * (blockDim.x >> 5) + (threadIdx.x >> 5);  // G/16 tiles
    const int n = lane & 15;
    const int hiHalf = lane >> 4;
    const int klo = hiHalf * 16;   // B: lanes 0-15 -> K 0..15, lanes 16-31 -> K 16..31

    BF16x16 b0, b1;
#pragma unroll
    for (int k = 0; k < 16; ++k) {
        int kk = klo + k;
        b0.u[k] = f2bf(Wl[kk * NC + n]);                       // cols 0..15, always < 26
        b1.u[k] = (16 + n < NC) ? f2bf(Wl[kk * NC + 16 + n]) : (unsigned short)0;
    }

    // A: full K=32. lanes 0-15: u[0..15]=K0..15; lanes 16-31: u[0..15]=K16..31
    const int row = tile * 16 + n;
    BF16x16 a;
    const unsigned short* ap = pooledbf + (size_t)row * 32;
#pragma unroll
    for (int q = 0; q < 8; ++q) {
        a.u[q]     = ap[hiHalf * 16 + q];
        a.u[8 + q] = ap[hiHalf * 16 + 8 + q];
    }

    v8f c0 = {}, c1 = {};
    c0 = __builtin_amdgcn_wmma_f32_16x16x32_bf16(false, a.v, false, b0.v,
                                                 (short)0, c0, false, false);
    c1 = __builtin_amdgcn_wmma_f32_16x16x32_bf16(false, a.v, false, b1.v,
                                                 (short)0, c1, false, false);
#pragma unroll
    for (int r = 0; r < 8; ++r) {
        const size_t orow = (size_t)tile * 16 + r + hiHalf * 8;
        out[orow * NC + n] = c0[r] + bl[n];
        if (16 + n < NC) out[orow * NC + 16 + n] = c1[r] + bl[16 + n];
    }
}

extern "C" void kernel_launch(void* const* d_in, const int* in_sizes, int n_in,
                              void* d_out, int out_size, void* d_ws, size_t ws_size,
                              hipStream_t stream) {
    const float* x   = (const float*)d_in[0];
    const int*   ei  = (const int*)d_in[1];
    const int*   src = ei;
    const int*   dst = ei + NE;
    const int*   batch = (const int*)d_in[2];
    const float* W1 = (const float*)d_in[3];
    const float* b1 = (const float*)d_in[4];
    const float* W2 = (const float*)d_in[5];
    const float* b2 = (const float*)d_in[6];
    const float* Wl = (const float*)d_in[7];
    const float* bl = (const float*)d_in[8];
    float* out = (float*)d_out;

    char* ws = (char*)d_ws;
    const size_t MB = 1ull << 20;
    float* dis   = (float*)(ws + 0);            // [0,1MB)   deg -> rsqrt in place
    float* h1lin = (float*)(ws + 1 * MB);       // [1,17)
    float* agg1  = (float*)(ws + 17 * MB);      // [17,33)
    float* h2lin = h1lin;                       // [1,33)    reuses h1lin+agg1 after post1
    unsigned short* h1bf = (unsigned short*)(ws + 33 * MB);   // [33,41)
    float* agg2  = (float*)(ws + 41 * MB);      // [41,73)   zero-filled inside k_gemm2
    float* pooled = (float*)(ws + 73 * MB);     // [73,75)
    float* cnt    = (float*)(ws + 75 * MB);     // 64KB
    unsigned short* pooledbf = (unsigned short*)(ws + 76 * MB);  // [76,77)

    hipMemsetAsync(dis,    0, (size_t)NN * 4,       stream);
    hipMemsetAsync(agg1,   0, (size_t)NN * 16 * 4,  stream);
    hipMemsetAsync(pooled, 0, (size_t)NG * 32 * 4,  stream);
    hipMemsetAsync(cnt,    0, (size_t)NG * 4,       stream);

    k_deg<<<NE / 256, 256, 0, stream>>>(dst, dis);
    k_dis<<<NN / 256, 256, 0, stream>>>(dis);
    k_xw1<<<NN / 256, 256, 0, stream>>>(x, W1, h1lin);
    k_scatter<4><<<(NE * 16) / 256, 256, 0, stream>>>(src, dst, dis, h1lin, agg1);
    k_post1<<<(NN * 16) / 256, 256, 0, stream>>>(agg1, h1lin, dis, b1, h1bf);
    // 16384 M-tiles / TPW(4) = 4096 waves = 512 blocks of 8 waves
    k_gemm2<<<(NN / 16 / TPW) / 8, 256, 0, stream>>>(h1bf, W2, h2lin, agg2);
    k_scatter<5><<<((size_t)NE * 32) / 256, 256, 0, stream>>>(src, dst, dis, h2lin, agg2);
    k_post2<<<(NN * 32) / 256, 256, 0, stream>>>(agg2, h2lin, dis, b2, batch, pooled, cnt);
    k_pooldiv<<<(NG * 32) / 256, 256, 0, stream>>>(pooled, cnt, pooledbf);
    // 1024 M-tiles = 1024 waves = 128 blocks
    k_cls<<<(NG / 16) / 8, 256, 0, stream>>>(pooledbf, Wl, bl, out);
}
